// MultiHeadAttention_84971632984649
// MI455X (gfx1250) — matrix operand order repaired and checked by reading, not compiled, not run on hardware
//
#include <hip/hip_runtime.h>
#include <hip/hip_bf16.h>

// CDNA5 (gfx1250) wave32 WMMA flash-attention + fused-bias output projection.
//   kernel 1: workgroup-cooperative flash attention. K/V 32x64-f32 tiles are
//             staged into LDS by the Tensor Data Mover (tensor_load_to_lds,
//             double-buffered, TENSORcnt-tracked, LDS pad_enable for bank
//             spread), then consumed by V_WMMA_F32_16X16X32_F16. Online
//             softmax with shfl_xor row reductions; P transposed D->A via LDS.
//   kernel 2: 4096x1024x1024 GEMM (context @ Wo^T + b), 2x2 WMMA tiles/wave.

typedef __attribute__((ext_vector_type(16))) _Float16 v16h;
typedef __attribute__((ext_vector_type(8)))  float    v8f;

#define HH 16
#define BB 2
#define TT 2048
#define DHD 64
#define MD 1024      // MODEL_DIM == HH*DHD
#define KB 32        // keys per staged tile
#define LSTR 65      // padded LDS row stride in dwords (TDM pad: +1 dw / 64 dw)

static __device__ __forceinline__ v8f wmma_f16(v16h a, v16h b, v8f c) {
  // D = A(16x32 f16) * B(32x16 f16) + C(16x16 f32)
  return __builtin_amdgcn_wmma_f32_16x16x32_f16(
      /*neg_a=*/false, a, /*neg_b=*/false, b,
      /*c_mod=*/(short)0, c, /*reuse_a=*/false, /*reuse_b=*/false);
}

// A-fragment (16x32, f16). Lane holds row = lane&15.
// elements 0..15 = K = c0..c0+15,
// c0 = 0 (lanes 0-15) or 16 (lanes 16-31). `row` points at K=0 of the row.
static __device__ __forceinline__ v16h a_frag_from_row(const float* __restrict__ row,
                                                       int c0, float scale) {
  v16h r;
#pragma unroll
  for (int e = 0; e < 8; ++e) {
    r[e]     = (_Float16)(row[c0 + e] * scale);
    r[e + 8] = (_Float16)(row[c0 + 8 + e] * scale);
  }
  return r;
}

// B-fragment (32x16, f16), K contiguous: lane holds N = lane&15, element e = K.
static __device__ __forceinline__ v16h b_frag_contig(const float* p) {
  v16h r;
#pragma unroll
  for (int e = 0; e < 16; ++e) r[e] = (_Float16)p[e];
  return r;
}

// B-fragment with strided K (V tile in LDS: K = key index, stride = LSTR).
static __device__ __forceinline__ v16h b_frag_strided(const float* p, int stride) {
  v16h r;
#pragma unroll
  for (int e = 0; e < 16; ++e) r[e] = (_Float16)p[e * stride];
  return r;
}

// Tensor Data Mover: DMA a 32(row) x 64(col) f32 tile (row stride 64 floats in
// global) into LDS with 1-dword padding per 64 dwords (LDS row stride = 65).
// D# packing per CDNA5 ISA. This toolchain's
// builtin takes the 6-arg form: (g0, g1, g2, g3, extra, cpol).
static __device__ __forceinline__ void tdm_load_tile(const float* gsrc, float* lds_dst) {
#if __has_builtin(__builtin_amdgcn_tensor_load_to_lds)
  typedef unsigned int v4u __attribute__((ext_vector_type(4)));
  typedef int v4i __attribute__((ext_vector_type(4)));
  typedef int v8i __attribute__((ext_vector_type(8)));
  const unsigned long long ga = (unsigned long long)(uintptr_t)gsrc;
  const unsigned lds = (unsigned)(uintptr_t)lds_dst;   // LDS aperture low bits = offset
  v4u g0;
  g0.x = 1u;                                            // count = 1 (valid D#)
  g0.y = lds;                                           // lds_addr (bytes)
  g0.z = (unsigned)ga;                                  // global_addr[31:0]
  g0.w = (unsigned)((ga >> 32) & 0x1FFFFFFull) | (2u << 30);  // addr[56:32], type=2
  v8i g1;
  g1[0] = (2 << 16) | (1 << 20) | (5 << 22);  // data_size=4B, pad_enable,
                                              // pad_interval=64dw, pad_amount=1dw
  g1[1] = (DHD & 0xFFFF) << 16;               // tensor_dim0 = 64
  g1[2] = (TT  & 0xFFFF) << 16;               // tensor_dim1 = 2048 (dim0 hi = 0)
  g1[3] = (DHD & 0xFFFF) << 16;               // tile_dim0 = 64 (dim1 hi = 0)
  g1[4] = KB;                                 // tile_dim1 = 32 (tile_dim2 = 0)
  g1[5] = DHD;                                // tensor_dim0_stride = 64
  g1[6] = 0;
  g1[7] = 0;
  const v4i gz4 = {0, 0, 0, 0};               // groups 2/3 unused (2D tensor)
  const v8i gz8 = {0, 0, 0, 0, 0, 0, 0, 0};
  __builtin_amdgcn_tensor_load_to_lds(g0, g1, gz4, gz4, gz8, /*cpol=*/0);
#else
  // Fallback: plain cooperative copy by the calling wave.
  const int lane = threadIdx.x & 31;
  for (int i = lane; i < KB * DHD; i += 32)
    lds_dst[(i / DHD) * LSTR + (i % DHD)] = gsrc[i];
#endif
}

static __device__ __forceinline__ void tdm_wait0() {
#if __has_builtin(__builtin_amdgcn_s_wait_tensorcnt)
  __builtin_amdgcn_s_wait_tensorcnt(0);
#else
  asm volatile("s_wait_tensorcnt 0" ::: "memory");
#endif
}

__global__ __launch_bounds__(128)
void fa_fwd_kernel(const float* __restrict__ q, const float* __restrict__ k,
                   const float* __restrict__ v, float* __restrict__ ctx) {
  __shared__ float   kbuf[2][KB * LSTR];   // double-buffered K tile (8.3 KB each)
  __shared__ float   vbuf[2][KB * LSTR];   // double-buffered V tile
  __shared__ _Float16 pbuf[4][16 * 32];    // per-wave P transpose staging

  const int lane = threadIdx.x & 31;
  const int wave = threadIdx.x >> 5;
  const int h = blockIdx.z, b = blockIdx.y;
  const int m0 = blockIdx.x * 64 + wave * 16;     // first query row of this wave

  const size_t hb = ((size_t)h * BB + b) * TT * DHD;
  const float* Q = q + hb;
  const float* K = k + hb;
  const float* V = v + hb;

  const int row_a = lane & 15;                 // A-layout row held by this lane
  const int c0    = (lane < 16) ? 0 : 16;      // A-layout K sub-offset
  const int colN  = lane & 15;                 // B/C/D-layout column
  const int rbase = (lane < 16) ? 0 : 8;       // C/D rows = rbase + r

  // Q fragments, pre-scaled by 1/sqrt(DH) = 0.125
  const float* qrow = Q + (size_t)(m0 + row_a) * DHD;
  const v16h aq0 = a_frag_from_row(qrow,      c0, 0.125f);
  const v16h aq1 = a_frag_from_row(qrow + 32, c0, 0.125f);

  v8f o0 = {}, o1 = {}, o2 = {}, o3 = {};
  float mrow[8], lrow[8];
#pragma unroll
  for (int r = 0; r < 8; ++r) { mrow[r] = -1e30f; lrow[r] = 0.0f; }

  _Float16* P = &pbuf[wave][0];
  const v8f zero = {};

  const int ntiles = blockIdx.x * 2 + 2;       // uniform per block (causal hi edge)

  // Prologue: DMA tile 0 (wave 0 issues; TDM ignores EXEC, one issue per wave)
  if (wave == 0) {
    tdm_load_tile(K, &kbuf[0][0]);
    tdm_load_tile(V, &vbuf[0][0]);
    tdm_wait0();
  }
  __syncthreads();

  for (int t = 0; t < ntiles; ++t) {
    const int j0  = t * KB;
    const int cur = t & 1;

    // Kick off next tile's DMA into the other buffer (overlaps WMMA below).
    if (wave == 0 && t + 1 < ntiles) {
      tdm_load_tile(K + (size_t)(j0 + KB) * DHD, &kbuf[cur ^ 1][0]);
      tdm_load_tile(V + (size_t)(j0 + KB) * DHD, &vbuf[cur ^ 1][0]);
    }

    if (j0 < m0 + 16) {   // causal: this wave still has unmasked keys here
      const float* kt = &kbuf[cur][0];
      const float* vt = &vbuf[cur][0];

      // ---- S = (Q/sqrt(dh)) K^T : two 16-key tiles, K-dim 64 split in 2 ----
      const float* kl = kt + (size_t)colN * LSTR + ((lane < 16) ? 0 : 16);
      const float* kr = kl + 16 * LSTR;
      v8f sL = wmma_f16(aq0, b_frag_contig(kl),      zero);
      sL     = wmma_f16(aq1, b_frag_contig(kl + 32), sL);
      v8f sR = wmma_f16(aq0, b_frag_contig(kr),      zero);
      sR     = wmma_f16(aq1, b_frag_contig(kr + 32), sR);

      // ---- causal mask (only blocks overlapping the diagonal) ----
      if (j0 + KB > m0) {
#pragma unroll
        for (int r = 0; r < 8; ++r) {
          const int rowg = m0 + rbase + r;
          if (j0 + colN      > rowg) sL[r] = -1e30f;
          if (j0 + 16 + colN > rowg) sR[r] = -1e30f;
        }
      }

      // ---- online softmax: each row lives in one 16-lane half ----
#pragma unroll
      for (int r = 0; r < 8; ++r) {
        float mx = fmaxf(sL[r], sR[r]);
        mx = fmaxf(mx, __shfl_xor(mx, 1, 32));
        mx = fmaxf(mx, __shfl_xor(mx, 2, 32));
        mx = fmaxf(mx, __shfl_xor(mx, 4, 32));
        mx = fmaxf(mx, __shfl_xor(mx, 8, 32));
        const float mnew = fmaxf(mrow[r], mx);
        const float pl = __expf(sL[r] - mnew);
        const float pr = __expf(sR[r] - mnew);
        float rs = pl + pr;
        rs += __shfl_xor(rs, 1, 32);
        rs += __shfl_xor(rs, 2, 32);
        rs += __shfl_xor(rs, 4, 32);
        rs += __shfl_xor(rs, 8, 32);
        const float alpha = __expf(mrow[r] - mnew);
        mrow[r] = mnew;
        lrow[r] = lrow[r] * alpha + rs;
        o0[r] *= alpha; o1[r] *= alpha; o2[r] *= alpha; o3[r] *= alpha;
        // stash P (D-layout) into LDS, row-major 16x32 f16
        P[(rbase + r) * 32 + colN]      = (_Float16)pl;
        P[(rbase + r) * 32 + 16 + colN] = (_Float16)pr;
      }
      asm volatile("s_wait_dscnt 0" ::: "memory");   // wave-local store->load order

      // ---- reload P in A-layout (D->A transpose via LDS) ----
      v16h pa;
#pragma unroll
      for (int e = 0; e < 8; ++e) {
        pa[e]     = P[row_a * 32 + c0 + e];
        pa[e + 8] = P[row_a * 32 + c0 + 8 + e];
      }

      // ---- O += P * V : K = 32 keys, four 16-wide DH tiles ----
      // LDS row stride 65 puts the two lane-halves on disjoint bank groups.
      const float* vb = vt + (size_t)((lane < 16) ? 0 : 16) * LSTR + colN;
      o0 = wmma_f16(pa, b_frag_strided(vb,      LSTR), o0);
      o1 = wmma_f16(pa, b_frag_strided(vb + 16, LSTR), o1);
      o2 = wmma_f16(pa, b_frag_strided(vb + 32, LSTR), o2);
      o3 = wmma_f16(pa, b_frag_strided(vb + 48, LSTR), o3);
    }

    __syncthreads();                               // all waves done with buf[cur]
    if (wave == 0 && t + 1 < ntiles) tdm_wait0();  // next buffer DMA complete
    __syncthreads();                               // publish to all waves
  }

  // ---- normalize and write context in [B, T, H*DH] layout ----
#pragma unroll
  for (int r = 0; r < 8; ++r) {
    const float inv = 1.0f / lrow[r];
    const size_t row = (size_t)b * TT + (size_t)(m0 + rbase + r);
    float* dst = ctx + row * MD + h * DHD;
    dst[colN]      = o0[r] * inv;
    dst[16 + colN] = o1[r] * inv;
    dst[32 + colN] = o2[r] * inv;
    dst[48 + colN] = o3[r] * inv;
  }
}

__global__ __launch_bounds__(128)
void proj_kernel(const float* __restrict__ ctx, const float* __restrict__ W,
                 const float* __restrict__ bias, float* __restrict__ out) {
  const int lane = threadIdx.x & 31;
  const int wave = threadIdx.x >> 5;
  const int rowbase = blockIdx.x * 64 + (wave & 1) * 32;
  const int colbase = blockIdx.y * 64 + (wave >> 1) * 32;
  const int row_a = lane & 15;
  const int c0    = (lane < 16) ? 0 : 16;
  const int colN  = lane & 15;
  const int rbase = (lane < 16) ? 0 : 8;

  v8f acc00 = {}, acc01 = {}, acc10 = {}, acc11 = {};

  for (int kb = 0; kb < MD; kb += 32) {
    const v16h a0 = a_frag_from_row(ctx + (size_t)(rowbase + row_a) * MD + kb, c0, 1.0f);
    const v16h a1 = a_frag_from_row(ctx + (size_t)(rowbase + 16 + row_a) * MD + kb, c0, 1.0f);
    // B[k][n] = W[n][k]; W rows are K-contiguous -> coalesced 16-float runs
    const float* w0 = W + (size_t)(colbase + colN) * MD + kb + ((lane < 16) ? 0 : 16);
    const float* w1 = w0 + 16 * MD;
    const v16h b0 = b_frag_contig(w0);
    const v16h b1 = b_frag_contig(w1);
    acc00 = wmma_f16(a0, b0, acc00);
    acc01 = wmma_f16(a0, b1, acc01);
    acc10 = wmma_f16(a1, b0, acc10);
    acc11 = wmma_f16(a1, b1, acc11);
  }

  const float bv0 = bias[colbase + colN];
  const float bv1 = bias[colbase + 16 + colN];
#pragma unroll
  for (int r = 0; r < 8; ++r) {
    const int r0 = rowbase + rbase + r;
    const int r1 = rowbase + 16 + rbase + r;
    out[(size_t)r0 * MD + colbase + colN]      = acc00[r] + bv0;
    out[(size_t)r0 * MD + colbase + 16 + colN] = acc01[r] + bv1;
    out[(size_t)r1 * MD + colbase + colN]      = acc10[r] + bv0;
    out[(size_t)r1 * MD + colbase + 16 + colN] = acc11[r] + bv1;
  }
}

extern "C" void kernel_launch(void* const* d_in, const int* in_sizes, int n_in,
                              void* d_out, int out_size, void* d_ws, size_t ws_size,
                              hipStream_t stream) {
  const float* q    = (const float*)d_in[0];
  const float* k    = (const float*)d_in[1];
  const float* v    = (const float*)d_in[2];
  const float* Wo_w = (const float*)d_in[3];
  const float* Wo_b = (const float*)d_in[4];
  float* out = (float*)d_out;
  float* ctx = (float*)d_ws;   // B*T*MD floats = 16 MB scratch

  dim3 g1(TT / 64, BB, HH);                 // 32 x 2 x 16 = 1024 blocks
  fa_fwd_kernel<<<g1, 128, 0, stream>>>(q, k, v, ctx);

  dim3 g2((BB * TT) / 64, MD / 64);         // 64 x 16 = 1024 blocks
  proj_kernel<<<g2, 128, 0, stream>>>(ctx, Wo_w, Wo_b, out);
}
